// GraphConvolution_23106924053066
// MI455X (gfx1250) — matrix operand order repaired and checked by reading, not compiled, not run on hardware
//
#include <hip/hip_runtime.h>

// ---------------------------------------------------------------------------
// GraphConvolution node-layer for MI455X (gfx1250, wave32, WMMA).
//
// ret = ((diag1(T·diag(d)·Tᵀ)) ⊙ adj_v) @ (H_v@W) + bias ; plus H_e passthrough
//
// Dominant GEMM: 4096x4096x8192 (~275 GFLOP) in v_wmma_f32_16x16x32_bf16 with
// fp32 accumulation. All LDS tiles are laid out so each 16x32 fragment is two
// ds_load_b128; staging packs bf16 pairs and writes b128. k_ret uses
// global_load_async_to_lds_b128 (ASYNCcnt) for its pure-bf16 tile copies.
// ---------------------------------------------------------------------------

typedef __attribute__((ext_vector_type(16))) __bf16        v16bf;
typedef __attribute__((ext_vector_type(8)))  float         v8f;
typedef __attribute__((ext_vector_type(4)))  unsigned int  v4u;

#define NV 4096
#define NE 8192
#define FV 128   // FV_IN == FV_OUT
#define FE 64

__device__ __forceinline__ unsigned short f32_to_bf16(float f) {
  unsigned int x = __float_as_uint(f);
  x += 0x7FFFu + ((x >> 16) & 1u);       // round-to-nearest-even
  return (unsigned short)(x >> 16);
}
__device__ __forceinline__ unsigned int pack2_bf16(float a, float b) {
  return (unsigned int)f32_to_bf16(a) | ((unsigned int)f32_to_bf16(b) << 16);
}

union FragBF { v4u q[2]; v16bf bf; };

// A fragment: 16x32 bf16 from LDS tile [m][ldk].
// Lane layout (ISA 7.12.2): M = lane%16, g = lane/16; K pairs are contiguous:
// VGPRs 0..3 cover K = g*8 + 0..7, VGPRs 4..7 cover K = 16 + g*8 + 0..7.
__device__ __forceinline__ v16bf frag_a(const unsigned short* s, int m0, int ldk,
                                        int k0, int lane) {
  const int m = m0 + (lane & 15);
  const int g = (lane >> 4) & 1;
  const unsigned short* base = &s[m * ldk + k0 + g * 8];
  FragBF f;
  f.q[0] = *(const v4u*)(base);          // 16B: K = g*8 + 0..7
  f.q[1] = *(const v4u*)(base + 16);     // 16B: K = 16 + g*8 + 0..7
  return f.bf;
}

// B fragment: 32x16 (KxN) from an LDS tile storing Bᵀ: [n][ldk].
// N = lane%16, g = lane/16; VGPRs 0..3 cover K = g*8 + 0..7, VGPRs 4..7 cover K = 16 + g*8 + 0..7.
__device__ __forceinline__ v16bf frag_b(const unsigned short* s, int n0, int ldk,
                                        int k0, int lane) {
  const int n = n0 + (lane & 15);
  const int g = (lane >> 4) & 1;
  const unsigned short* base = &s[n * ldk + k0 + g * 8];
  FragBF f;
  f.q[0] = *(const v4u*)(base);          // K = g*8 + 0..7
  f.q[1] = *(const v4u*)(base + 16);     // K = 16 + g*8 + 0..7
  return f.bf;
}

__device__ __forceinline__ v8f wmma_bf16(v16bf a, v16bf b, v8f c) {
  return __builtin_amdgcn_wmma_f32_16x16x32_bf16(false, a, false, b,
                                                 (short)0, c, false, false);
}

// ---------------------------------------------------------------------------
// Kernel 1: d[e] = H_e[e,:] . p  (one wave per edge) + H_e passthrough copy.
// ---------------------------------------------------------------------------
__global__ void k_edge_scalar(const float* __restrict__ He, const float* __restrict__ p,
                              float* __restrict__ d, float* __restrict__ outHe) {
  const int e    = blockIdx.x * 8 + (threadIdx.x >> 5);
  const int lane = threadIdx.x & 31;
  const float h0 = He[(size_t)e * FE + lane];
  const float h1 = He[(size_t)e * FE + 32 + lane];
  outHe[(size_t)e * FE + lane]      = h0;
  outHe[(size_t)e * FE + 32 + lane] = h1;
  float v = h0 * p[lane] + h1 * p[32 + lane];
#pragma unroll
  for (int off = 16; off > 0; off >>= 1) v += __shfl_xor(v, off, 32);
  if (lane == 0) d[e] = v;
}

// ---------------------------------------------------------------------------
// Kernel 2: Hw = H_v @ W  (4096x128x128), emitted TRANSPOSED as bf16:
// HwT[n][m], n=0..127, m=0..4095 — so k_ret can stage B tiles with b128 copies.
// Wave w owns a 32x64 sub-tile: mw = w&3 (rows 32mw..), nw = w>>2 (cols 64nw..).
// ---------------------------------------------------------------------------
__global__ void k_hv_weight(const float* __restrict__ Hv, const float* __restrict__ W,
                            unsigned short* __restrict__ HwT) {
  __shared__ unsigned short sWt[FV * FV];  // Wᵀ: [n][k]
  __shared__ unsigned short sH[128 * FV];  // H_v tile: [m][k]
  const int t  = threadIdx.x;
  const int i0 = blockIdx.x * 128;

  // stage Wᵀ (transpose during conversion) and H_v tile, packed dword stores
  for (int u = t; u < (FV * FV) / 2; u += 256) {
    const int n = (2 * u) >> 7, k = (2 * u) & 127;
    ((unsigned int*)sWt)[u] = pack2_bf16(W[(size_t)k * FV + n], W[(size_t)(k + 1) * FV + n]);
  }
  for (int u = t; u < (128 * FV) / 2; u += 256) {
    const int r = (2 * u) >> 7, c = (2 * u) & 127;
    const float2 hv = *(const float2*)&Hv[(size_t)(i0 + r) * FV + c];
    ((unsigned int*)sH)[u] = pack2_bf16(hv.x, hv.y);
  }
  __syncthreads();

  const int wave = t >> 5, lane = t & 31;
  const int mw = (wave & 3) * 32, nw = (wave >> 2) * 64;
  v8f acc[2][4] = {};
#pragma unroll
  for (int ks = 0; ks < 4; ++ks) {
    v16bf a0 = frag_a(sH, mw, FV, ks * 32, lane);
    v16bf a1 = frag_a(sH, mw + 16, FV, ks * 32, lane);
#pragma unroll
    for (int nt = 0; nt < 4; ++nt) {
      const v16bf b = frag_b(sWt, nw + nt * 16, FV, ks * 32, lane);
      acc[0][nt] = wmma_bf16(a0, b, acc[0][nt]);
      acc[1][nt] = wmma_bf16(a1, b, acc[1][nt]);
    }
  }
  const int g = (lane >> 4) & 1, n16 = lane & 15;
#pragma unroll
  for (int ai = 0; ai < 2; ++ai)
#pragma unroll
    for (int nt = 0; nt < 4; ++nt)
#pragma unroll
      for (int v = 0; v < 8; ++v) {
        const int gi = i0 + mw + ai * 16 + v + 8 * g;   // C/D layout: M = v + 8g
        const int gn = nw + nt * 16 + n16;
        HwT[(size_t)gn * NV + gi] = f32_to_bf16(acc[ai][nt][v]);
      }
}

// ---------------------------------------------------------------------------
// Kernel 3 (dominant): S = (T·diag(d)) @ Tᵀ per 128x128 tile, K=8192;
// then adjA = mask(S) ⊙ adj_v, stored bf16. 32x32 workgroups, 8 waves each,
// wave = 32x64 sub-tile (2 A-frags + 4 B-frags -> 8 WMMA per K-step).
// ---------------------------------------------------------------------------
__global__ void k_multiplier(const float* __restrict__ T, const float* __restrict__ d,
                             const float* __restrict__ adjv,
                             unsigned short* __restrict__ adjA) {
  __shared__ unsigned short sA[128 * 32];  // (T⊙d) rows i0.., [m][k]
  __shared__ unsigned short sB[128 * 32];  // T rows j0..,     [n][k]  (= Bᵀ)
  const int t  = threadIdx.x;
  const int i0 = blockIdx.y * 128;
  const int j0 = blockIdx.x * 128;
  const int wave = t >> 5, lane = t & 31;
  const int mw = (wave & 3) * 32, nw = (wave >> 2) * 64;
  const int row = t >> 1;                 // 0..127: staging row
  const int cB  = (t & 1) * 16;           // 0 or 16: staging column half

  v8f acc[2][4] = {};
  const float* __restrict__ rowA = T + (size_t)(i0 + row) * NE;
  const float* __restrict__ rowB = T + (size_t)(j0 + row) * NE;

  for (int k0 = 0; k0 < NE; k0 += 32) {
    // stage 16 elements per thread per tile; convert+pack, b128 stores
#pragma unroll
    for (int h = 0; h < 2; ++h) {
      const int c = k0 + cB + h * 8;
      const float4 ta0 = *(const float4*)&rowA[c];
      const float4 ta1 = *(const float4*)&rowA[c + 4];
      const float4 dv0 = *(const float4*)&d[c];
      const float4 dv1 = *(const float4*)&d[c + 4];
      v4u ua;
      ua[0] = pack2_bf16(ta0.x * dv0.x, ta0.y * dv0.y);
      ua[1] = pack2_bf16(ta0.z * dv0.z, ta0.w * dv0.w);
      ua[2] = pack2_bf16(ta1.x * dv1.x, ta1.y * dv1.y);
      ua[3] = pack2_bf16(ta1.z * dv1.z, ta1.w * dv1.w);
      *(v4u*)&sA[row * 32 + cB + h * 8] = ua;
      const float4 tb0 = *(const float4*)&rowB[c];
      const float4 tb1 = *(const float4*)&rowB[c + 4];
      v4u ub;
      ub[0] = pack2_bf16(tb0.x, tb0.y);
      ub[1] = pack2_bf16(tb0.z, tb0.w);
      ub[2] = pack2_bf16(tb1.x, tb1.y);
      ub[3] = pack2_bf16(tb1.z, tb1.w);
      *(v4u*)&sB[row * 32 + cB + h * 8] = ub;
    }
    if (k0 + 32 < NE) {                    // global_prefetch_b8 for next K tile
      __builtin_prefetch(&rowA[k0 + 32 + cB], 0, 1);
      __builtin_prefetch(&rowB[k0 + 32 + cB], 0, 1);
    }
    __syncthreads();
    {
      const v16bf a0 = frag_a(sA, mw, 32, 0, lane);
      const v16bf a1 = frag_a(sA, mw + 16, 32, 0, lane);
#pragma unroll
      for (int nt = 0; nt < 4; ++nt) {
        const v16bf b = frag_b(sB, nw + nt * 16, 32, 0, lane);
        acc[0][nt] = wmma_bf16(a0, b, acc[0][nt]);
        acc[1][nt] = wmma_bf16(a1, b, acc[1][nt]);
      }
    }
    __syncthreads();
  }

  // diag -> 1, elementwise * adj_v, emit bf16
  const int g = (lane >> 4) & 1, n16 = lane & 15;
#pragma unroll
  for (int ai = 0; ai < 2; ++ai)
#pragma unroll
    for (int nt = 0; nt < 4; ++nt) {
      const int gj = j0 + nw + nt * 16 + n16;
#pragma unroll
      for (int v = 0; v < 8; ++v) {
        const int gi = i0 + mw + ai * 16 + v + 8 * g;
        float val = acc[ai][nt][v];
        if (gi == gj) val = 1.0f;
        val *= adjv[(size_t)gi * NV + gj];
        adjA[(size_t)gi * NV + gj] = f32_to_bf16(val);
      }
    }
}

// ---------------------------------------------------------------------------
// Kernel 4: ret = adjA(bf16) @ Hw(bf16) + bias  (4096x128, K=4096), fp32 out.
// Tiles are already bf16 -> stage them with global_load_async_to_lds_b128
// (ASYNCcnt path, no VGPR round-trip), then s_wait_asynccnt + barrier.
// ---------------------------------------------------------------------------
__global__ void k_ret(const unsigned short* __restrict__ adjA,
                      const unsigned short* __restrict__ HwT,
                      const float* __restrict__ bias, float* __restrict__ out) {
  __shared__ unsigned short sA[128 * 32];  // adjA tile [m][k]
  __shared__ unsigned short sB[128 * 32];  // HwT tile  [n][k]  (= Bᵀ)
  const int t  = threadIdx.x;
  const int i0 = blockIdx.x * 128;
  const int wave = t >> 5, lane = t & 31;
  const int mw = (wave & 3) * 32, nw = (wave >> 2) * 64;
  const int row = t >> 1;
  const int cB  = (t & 1) * 16;

  // flat->LDS: low 32 bits of a flat LDS address are the LDS byte address
  const unsigned int ldsA = (unsigned int)(size_t)&sA[row * 32 + cB];
  const unsigned int ldsB = (unsigned int)(size_t)&sB[row * 32 + cB];

  v8f acc[2][4] = {};
  for (int k0 = 0; k0 < NV; k0 += 32) {
    {
      const unsigned long long gA =
          (unsigned long long)&adjA[(size_t)(i0 + row) * NV + k0 + cB];
      const unsigned long long gB =
          (unsigned long long)&HwT[(size_t)row * NV + k0 + cB];
      asm volatile("global_load_async_to_lds_b128 %0, %1, off"
                   :: "v"(ldsA), "v"(gA) : "memory");
      asm volatile("global_load_async_to_lds_b128 %0, %1, off offset:16"
                   :: "v"(ldsA), "v"(gA) : "memory");
      asm volatile("global_load_async_to_lds_b128 %0, %1, off"
                   :: "v"(ldsB), "v"(gB) : "memory");
      asm volatile("global_load_async_to_lds_b128 %0, %1, off offset:16"
                   :: "v"(ldsB), "v"(gB) : "memory");
      asm volatile("s_wait_asynccnt 0" ::: "memory");
    }
    __syncthreads();
    {
      const v16bf a0 = frag_a(sA, mw, 32, 0, lane);
      const v16bf a1 = frag_a(sA, mw + 16, 32, 0, lane);
#pragma unroll
      for (int nt = 0; nt < 4; ++nt) {
        const v16bf b = frag_b(sB, nw + nt * 16, 32, 0, lane);
        acc[0][nt] = wmma_bf16(a0, b, acc[0][nt]);
        acc[1][nt] = wmma_bf16(a1, b, acc[1][nt]);
      }
    }
    __syncthreads();
  }

  const int g = (lane >> 4) & 1, n16 = lane & 15;
#pragma unroll
  for (int ai = 0; ai < 2; ++ai)
#pragma unroll
    for (int nt = 0; nt < 4; ++nt) {
      const int gn = nw + nt * 16 + n16;
      const float b = bias[gn];
#pragma unroll
      for (int v = 0; v < 8; ++v) {
        const int gi = i0 + mw + ai * 16 + v + 8 * g;
        out[(size_t)gi * FV + gn] = acc[ai][nt][v] + b;
      }
    }
}

// ---------------------------------------------------------------------------
extern "C" void kernel_launch(void* const* d_in, const int* in_sizes, int n_in,
                              void* d_out, int out_size, void* d_ws, size_t ws_size,
                              hipStream_t stream) {
  (void)in_sizes; (void)n_in; (void)out_size; (void)ws_size;

  const float* Hv   = (const float*)d_in[0];
  const float* He   = (const float*)d_in[1];
  /* d_in[2] = adj_e (unused by node-layer branch) */
  const float* adjv = (const float*)d_in[3];
  const float* T    = (const float*)d_in[4];
  const float* W    = (const float*)d_in[5];
  const float* p    = (const float*)d_in[6];
  const float* bias = (const float*)d_in[7];

  float* out_ret = (float*)d_out;
  float* out_He  = out_ret + (size_t)NV * FV;

  // workspace layout: d (32 KB) | HwT bf16 (1 MB) | adjA bf16 (32 MB)
  char* ws = (char*)d_ws;
  float*          dvec = (float*)ws;
  unsigned short* HwT  = (unsigned short*)(ws + 64 * 1024);
  unsigned short* adjA = (unsigned short*)(ws + 64 * 1024 + (size_t)NV * FV * 2);

  k_edge_scalar<<<dim3(NE / 8),             dim3(256), 0, stream>>>(He, p, dvec, out_He);
  k_hv_weight  <<<dim3(NV / 128),           dim3(256), 0, stream>>>(Hv, W, HwT);
  k_multiplier <<<dim3(NV / 128, NV / 128), dim3(256), 0, stream>>>(T, dvec, adjv, adjA);
  k_ret        <<<dim3(NV / 128),           dim3(256), 0, stream>>>(adjA, HwT, bias, out_ret);
}
